// ConceptMoE_2224793059973
// MI455X (gfx1250) — matrix operand order repaired and checked by reading, not compiled, not run on hardware
//
#include <hip/hip_runtime.h>
#include <hip/hip_bf16.h>

// ---------------------------------------------------------------------------
// ConceptMoE forward for MI455X (gfx1250, wave32, WMMA + TDM/async-LDS).
//   h   = emb[ids] * 16
//   Q   = h[:-1] @ q_w^T ; K = h[1:] @ k_w^T     (bf16 WMMA, f32 accum)
//   bp/mask/sel from cosine sim, aux_loss reduction
//   EMA scan over boundaries -> h_final = (h + state) * 16   (bf16)
//   logits = h_final @ lm_w^T                     (bf16 WMMA, f32 accum)
// lm_w pre-converted to bf16 (131 MB) -> L2-resident (192 MB global L2).
// GEMM tiles double-buffered in LDS, staged with TDM (tensor_load_to_lds)
// for B and async global->LDS DMA for A, overlapping copy with WMMA.
// ---------------------------------------------------------------------------

typedef unsigned short u16;
typedef unsigned int   u32;
typedef unsigned long long u64;
typedef __attribute__((ext_vector_type(16))) __bf16 v16bf;
typedef __attribute__((ext_vector_type(8)))  float  v8f;
typedef __attribute__((ext_vector_type(4)))  int    v4i;

#define L_TOK 4096
#define HID   2048
#define VOCABN 32000
#define ENC_SCALE 16.0f
#define DEC_SCALE 16.0f
#define CPT_SCALE 70368744177664.0f   /* 4^23 = 2^46 */
#define BK 32

// ---- CDNA5 data-mover feature detection -----------------------------------
#if defined(__AMDGCN__)
#if __has_builtin(__builtin_amdgcn_tensor_load_to_lds)
#define HAVE_TDM 1
#endif
#if __has_builtin(__builtin_amdgcn_global_load_async_to_lds_b128)
#define HAVE_ASYNC 1
#endif
#endif

// AS1 = global ("__device__"), AS3 = LDS ("__shared__") per hipcc diagnostics.
typedef __attribute__((address_space(1))) v4i gv4i;
typedef __attribute__((address_space(3))) v4i lv4i;

// Low 32 bits of a generic LDS pointer are the LDS byte offset (ISA 10.2:
// LDS offset is address bits 31:0); integer round-trip avoids addrspacecast syntax.
__device__ __forceinline__ u32 lds_off(const void* p) {
    return (u32)(uintptr_t)p;
}
__device__ __forceinline__ gv4i* to_glob4(const void* p) {
    return (gv4i*)(u64)(uintptr_t)p;
}
__device__ __forceinline__ lv4i* to_lds4(void* p) {
    return (lv4i*)(u32)(uintptr_t)p;
}

#ifdef HAVE_TDM
typedef __attribute__((ext_vector_type(4))) unsigned int u32x4;
typedef __attribute__((ext_vector_type(8))) int i32x8;
typedef __attribute__((ext_vector_type(4))) int i32x4;

// 2D tile load (tile_rows x 32 bf16) via Tensor Data Mover, ISA §8 descriptor.
__device__ __forceinline__ void tdm_load_2d(const u16* gsrc, void* ldst,
                                            int tile_rows, int row_stride,
                                            int tensor_rows, int tensor_cols) {
    u64 ga = (u64)(uintptr_t)gsrc;
    u32x4 g0;
    g0[0] = 1u;                                              // count=1 (user)
    g0[1] = lds_off(ldst);                                   // lds_addr
    g0[2] = (u32)ga;                                         // global_addr lo
    g0[3] = (u32)((ga >> 32) & 0x01FFFFFFu) | (2u << 30);    // addr hi | type=2
    i32x8 g1;
    g1[0] = (int)(1u << 16);                                 // data_size=2B
    g1[1] = (int)(((u32)tensor_cols & 0xFFFFu) << 16);       // tensor_dim0 lo16
    g1[2] = (int)((((u32)tensor_cols >> 16) & 0xFFFFu) |
                  (((u32)tensor_rows & 0xFFFFu) << 16));     // dim0 hi|dim1 lo
    g1[3] = (int)((((u32)tensor_rows >> 16) & 0xFFFFu) |
                  ((u32)BK << 16));                          // dim1 hi|tile_dim0
    g1[4] = (int)(u32)tile_rows;                             // tile_dim1
    g1[5] = (int)(u32)row_stride;                            // dim0_stride lo32
    g1[6] = 0;
    g1[7] = 0;
    i32x4 z4 = {0, 0, 0, 0};
#if __clang_major__ >= 23
    i32x8 z8 = {0, 0, 0, 0, 0, 0, 0, 0};
    __builtin_amdgcn_tensor_load_to_lds(g0, g1, z4, z4, z8, 0);
#else
    __builtin_amdgcn_tensor_load_to_lds(g0, g1, z4, z4, 0);
#endif
}
#endif  // HAVE_TDM

__device__ __forceinline__ u16 f2bf(float f) {
    union { float f; u32 u; } x; x.f = f;
    u32 r = x.u + 0x7FFFu + ((x.u >> 16) & 1u);   // round-to-nearest-even
    return (u16)(r >> 16);
}

// ---------------- gather + scale: h = emb[ids]*16, bf16 copy (+pad row) ----
__global__ __launch_bounds__(256) void gather_scale_kernel(
    const int* __restrict__ ids, const float* __restrict__ emb,
    float* __restrict__ h, u16* __restrict__ hbf) {
    int t = blockIdx.x;
    if (t == L_TOK) {                       // zero pad row for shifted K-proj
        u32* dst = (u32*)(hbf + (size_t)L_TOK * HID);
        for (int j = threadIdx.x; j < HID / 2; j += 256) dst[j] = 0u;
        return;
    }
    int id = ids[t];
    const float4* src = (const float4*)(emb + (size_t)id * HID);
    float4* hdst = (float4*)(h + (size_t)t * HID);
    uint2*  bdst = (uint2*)(hbf + (size_t)t * HID);
    for (int j4 = threadIdx.x; j4 < HID / 4; j4 += 256) {
        float4 v = src[j4];
        v.x *= ENC_SCALE; v.y *= ENC_SCALE; v.z *= ENC_SCALE; v.w *= ENC_SCALE;
        hdst[j4] = v;
        uint2 o;
        o.x = (u32)f2bf(v.x) | ((u32)f2bf(v.y) << 16);
        o.y = (u32)f2bf(v.z) | ((u32)f2bf(v.w) << 16);
        bdst[j4] = o;
    }
}

// ---------------- f32 -> bf16 bulk convert ---------------------------------
__global__ __launch_bounds__(256) void cvt_bf16_kernel(
    const float* __restrict__ src, u16* __restrict__ dst, int n4) {
    for (int i = blockIdx.x * 256 + threadIdx.x; i < n4; i += gridDim.x * 256) {
        float4 v = ((const float4*)src)[i];
        uint2 o;
        o.x = (u32)f2bf(v.x) | ((u32)f2bf(v.y) << 16);
        o.y = (u32)f2bf(v.z) | ((u32)f2bf(v.w) << 16);
        ((uint2*)dst)[i] = o;
    }
}

// ---------------- bf16 GEMM: C[M,N] = A[M,K] * B[N,K]^T --------------------
// 128x128 C-tile/block, 8 waves, wave w -> 16-row strip, 8 v8f accumulators.
// Double-buffered LDS tiles; staging via TDM / async-DMA when available.
__global__ __launch_bounds__(256) void gemm_bf16_nt_kernel(
    const u16* __restrict__ A, const u16* __restrict__ B, float* __restrict__ C,
    int Md, int Nd, int Kd, int lda, int ldb, int ldc) {
    __shared__ __align__(16) u16 As[2][128 * BK];
    __shared__ __align__(16) u16 Bs[2][128 * BK];

    const int t    = threadIdx.x;
    const int lane = t & 31;
    const int w    = t >> 5;
    const int half = lane >> 4;
    const int lrow = lane & 15;
    const size_t m0 = (size_t)blockIdx.y * 128;
    const size_t n0 = (size_t)blockIdx.x * 128;
    const int srow = t >> 1;             // 0..127   (staging row)
    const int sc   = (t & 1) * 16;       // 0 or 16  (staging col chunk)
    (void)srow; (void)sc; (void)Md; (void)Nd;

    v8f acc[8];
    #pragma unroll
    for (int n = 0; n < 8; ++n)
        acc[n] = (v8f){0.f, 0.f, 0.f, 0.f, 0.f, 0.f, 0.f, 0.f};

#if defined(HAVE_TDM) && defined(HAVE_ASYNC)
    // A tile: per-lane async DMA to LDS.  B tile: one TDM descriptor / block.
    auto stage = [&](int kk, int buf) {
        const u16* ga = A + (m0 + srow) * (size_t)lda + kk + sc;
        u16* la = &As[buf][srow * BK + sc];
        __builtin_amdgcn_global_load_async_to_lds_b128(to_glob4(ga),     to_lds4(la),     0, 0);
        __builtin_amdgcn_global_load_async_to_lds_b128(to_glob4(ga + 8), to_lds4(la + 8), 0, 0);
        if (t < 32)
            tdm_load_2d(B + n0 * (size_t)ldb + kk, &Bs[buf][0], 128, ldb, Nd, Kd - kk);
    };
    auto stage_wait = [&] {
        asm volatile("s_wait_asynccnt 0x0" ::: "memory");
        if (t < 32) asm volatile("s_wait_tensorcnt 0x0" ::: "memory");
    };
#elif defined(HAVE_ASYNC)
    auto stage = [&](int kk, int buf) {
        const u16* ga = A + (m0 + srow) * (size_t)lda + kk + sc;
        const u16* gb = B + (n0 + srow) * (size_t)ldb + kk + sc;
        u16* la = &As[buf][srow * BK + sc];
        u16* lb = &Bs[buf][srow * BK + sc];
        __builtin_amdgcn_global_load_async_to_lds_b128(to_glob4(ga),     to_lds4(la),     0, 0);
        __builtin_amdgcn_global_load_async_to_lds_b128(to_glob4(ga + 8), to_lds4(la + 8), 0, 0);
        __builtin_amdgcn_global_load_async_to_lds_b128(to_glob4(gb),     to_lds4(lb),     0, 0);
        __builtin_amdgcn_global_load_async_to_lds_b128(to_glob4(gb + 8), to_lds4(lb + 8), 0, 0);
    };
    auto stage_wait = [&] {
        asm volatile("s_wait_asynccnt 0x0" ::: "memory");
    };
#elif defined(HAVE_TDM)
    auto stage = [&](int kk, int buf) {
        if (t < 32) {
            tdm_load_2d(A + m0 * (size_t)lda + kk, &As[buf][0], 128, lda, Md, Kd - kk);
            tdm_load_2d(B + n0 * (size_t)ldb + kk, &Bs[buf][0], 128, ldb, Nd, Kd - kk);
        }
    };
    auto stage_wait = [&] {
        if (t < 32) asm volatile("s_wait_tensorcnt 0x0" ::: "memory");
    };
#else
    auto stage = [&](int kk, int buf) {
        const uint4* ga = (const uint4*)(A + (m0 + srow) * (size_t)lda + kk + sc);
        const uint4* gb = (const uint4*)(B + (n0 + srow) * (size_t)ldb + kk + sc);
        uint4 r0 = ga[0], r1 = ga[1], r2 = gb[0], r3 = gb[1];
        uint4* la = (uint4*)&As[buf][srow * BK + sc];
        uint4* lb = (uint4*)&Bs[buf][srow * BK + sc];
        la[0] = r0; la[1] = r1; lb[0] = r2; lb[1] = r3;
    };
    auto stage_wait = [&] {};
#endif

    stage(0, 0);
    int buf = 0;
    for (int kk = 0; kk < Kd; kk += BK) {
        stage_wait();
        __syncthreads();                       // tile `buf` visible to all
        if (kk + BK < Kd) stage(kk + BK, buf ^ 1);   // overlap DMA with WMMA

        // A fragment: row = 16w + lrow ; K = {0..7,16..23} + 8*half
        union { v16bf v; uint4 q[2]; } afr;
        const u16* ap = &As[buf][(16 * w + lrow) * BK];
        afr.q[0] = *(const uint4*)(ap + half * 8);
        afr.q[1] = *(const uint4*)(ap + 16 + half * 8);

        #pragma unroll
        for (int n = 0; n < 8; ++n) {          // 8 WMMAs reuse the A fragment
            union { v16bf v; uint4 q[2]; } bfr;
            const u16* bpp = &Bs[buf][(16 * n + lrow) * BK + half * 8];
            bfr.q[0] = *(const uint4*)(bpp);
            bfr.q[1] = *(const uint4*)(bpp + 16);
            acc[n] = __builtin_amdgcn_wmma_f32_16x16x32_bf16(
                false, afr.v, false, bfr.v, (short)0, acc[n], false, false);
        }
        buf ^= 1;
    }

    // epilogue: C/D layout -> lane half selects M+8, lrow = N
    #pragma unroll
    for (int n = 0; n < 8; ++n) {
        #pragma unroll
        for (int r = 0; r < 8; ++r) {
            size_t row = m0 + 16 * w + r + half * 8;
            size_t col = n0 + 16 * n + lrow;
            C[row * (size_t)ldc + col] = acc[n][r];
        }
    }
}

// ---------------- cosine / boundary prob / mask / sel ----------------------
__global__ __launch_bounds__(256) void chunk_kernel(
    const float* __restrict__ Q, const float* __restrict__ Kb,
    float* __restrict__ bp, float* __restrict__ sel, float* __restrict__ mk) {
    __shared__ float s0[256], s1[256], s2[256];
    int t = blockIdx.x;                    // 0..4094
    const float* q = Q  + (size_t)t * HID;
    const float* k = Kb + (size_t)t * HID; // Kb row t == proj of h[t+1]
    float qk = 0.f, qq = 0.f, kk = 0.f;
    for (int j = threadIdx.x; j < HID; j += 256) {
        float a = q[j], b = k[j];
        qk += a * b; qq += a * a; kk += b * b;
    }
    s0[threadIdx.x] = qk; s1[threadIdx.x] = qq; s2[threadIdx.x] = kk;
    __syncthreads();
    for (int o = 128; o > 0; o >>= 1) {
        if (threadIdx.x < o) {
            s0[threadIdx.x] += s0[threadIdx.x + o];
            s1[threadIdx.x] += s1[threadIdx.x + o];
            s2[threadIdx.x] += s2[threadIdx.x + o];
        }
        __syncthreads();
    }
    if (threadIdx.x == 0) {
        float cs = s0[0] / (fmaxf(sqrtf(s1[0]), 1e-12f) *
                            fmaxf(sqrtf(s2[0]), 1e-12f));
        float p  = fminf(fmaxf((1.f - cs) * 0.5f, 0.f), 1.f);
        bool  m  = p >= 0.5f;
        bp[t + 1]  = p;
        mk[t + 1]  = m ? 1.f : 0.f;
        sel[t + 1] = m ? p : 1.f - p;
        if (t == 0) { bp[0] = 1.f; mk[0] = 1.f; sel[0] = 1.f; }
    }
}

// ---------------- aux loss reduction (R = 2) -------------------------------
__global__ __launch_bounds__(256) void aux_kernel(
    const float* __restrict__ bp, const float* __restrict__ mk,
    float* __restrict__ out_aux) {
    __shared__ float s0[256], s1[256];
    float a = 0.f, b = 0.f;
    for (int i = threadIdx.x; i < L_TOK; i += 256) { a += bp[i]; b += mk[i]; }
    s0[threadIdx.x] = a; s1[threadIdx.x] = b;
    __syncthreads();
    for (int o = 128; o > 0; o >>= 1) {
        if (threadIdx.x < o) {
            s0[threadIdx.x] += s0[threadIdx.x + o];
            s1[threadIdx.x] += s1[threadIdx.x + o];
        }
        __syncthreads();
    }
    if (threadIdx.x == 0) {
        float G = s0[0] / (float)L_TOK;
        float F = s1[0] / (float)L_TOK;
        out_aux[0] = 2.f * (F * G + (1.f - F) * (1.f - G));  // R = 2
    }
}

// ---------------- boundary EMA scan + plug-back, bf16 output ---------------
__global__ __launch_bounds__(256) void scan_kernel(
    const float* __restrict__ h, const float* __restrict__ sel,
    const float* __restrict__ mk, u16* __restrict__ hout) {
    __shared__ float sp[L_TOK];
    __shared__ float sm[L_TOK];
    for (int i = threadIdx.x; i < L_TOK; i += 256) { sp[i] = sel[i]; sm[i] = mk[i]; }
    __syncthreads();
    int j = blockIdx.x * 256 + threadIdx.x;   // 8 blocks cover HID
    float state = 0.f;
    for (int t = 0; t < L_TOK; ++t) {
        float x = h[(size_t)t * HID + j];
        float p = sp[t];
        if (sm[t] != 0.f) state = state * (1.f - p) + x * CPT_SCALE * p;
        hout[(size_t)t * HID + j] = f2bf((x + state) * DEC_SCALE);
    }
}

// ---------------------------------------------------------------------------
extern "C" void kernel_launch(void* const* d_in, const int* in_sizes, int n_in,
                              void* d_out, int out_size, void* d_ws, size_t ws_size,
                              hipStream_t stream) {
    const int*   ids  = (const int*)  d_in[0];
    const float* emb  = (const float*)d_in[1];
    const float* q_w  = (const float*)d_in[2];
    const float* k_w  = (const float*)d_in[3];
    const float* lm_w = (const float*)d_in[4];
    float* out = (float*)d_out;

    // -------- workspace carve-out (256B aligned) --------
    char* ws = (char*)d_ws;
    size_t off = 0;
    auto carve = [&](size_t bytes) {
        size_t o = off; off = (off + bytes + 255) & ~(size_t)255; return o;
    };
    float* h      = (float*)(ws + carve((size_t)L_TOK * HID * 4));
    u16*   hbf    = (u16*)  (ws + carve((size_t)(L_TOK + 1) * HID * 2)); // +pad row
    u16*   qwbf   = (u16*)  (ws + carve((size_t)HID * HID * 2));
    u16*   kwbf   = (u16*)  (ws + carve((size_t)HID * HID * 2));
    u16*   lmwbf  = (u16*)  (ws + carve((size_t)VOCABN * HID * 2));
    float* Qbuf   = (float*)(ws + carve((size_t)L_TOK * HID * 4));
    float* Kbuf   = (float*)(ws + carve((size_t)L_TOK * HID * 4));
    u16*   hfbf   = (u16*)  (ws + carve((size_t)L_TOK * HID * 2));
    float* bpv    = (float*)(ws + carve((size_t)L_TOK * 4));
    float* selv   = (float*)(ws + carve((size_t)L_TOK * 4));
    float* mkv    = (float*)(ws + carve((size_t)L_TOK * 4));
    (void)ws_size; (void)in_sizes; (void)n_in; (void)out_size;

    // 1) gather + scale (+pad row)
    gather_scale_kernel<<<L_TOK + 1, 256, 0, stream>>>(ids, emb, h, hbf);

    // 2) weight conversions to bf16 (lm_w bf16 = 131 MB -> L2-resident)
    cvt_bf16_kernel<<<1024, 256, 0, stream>>>(q_w,  qwbf, HID * HID / 4);
    cvt_bf16_kernel<<<1024, 256, 0, stream>>>(k_w,  kwbf, HID * HID / 4);
    cvt_bf16_kernel<<<8192, 256, 0, stream>>>(lm_w, lmwbf, VOCABN * HID / 4);

    // 3) Q = h @ q_w^T ; K = h[1:] @ k_w^T  (shifted A, pad row keeps it legal)
    dim3 gqk(HID / 128, L_TOK / 128);
    gemm_bf16_nt_kernel<<<gqk, 256, 0, stream>>>(hbf,       qwbf, Qbuf,
                                                 L_TOK, HID, HID, HID, HID, HID);
    gemm_bf16_nt_kernel<<<gqk, 256, 0, stream>>>(hbf + HID, kwbf, Kbuf,
                                                 L_TOK, HID, HID, HID, HID, HID);

    // 4) boundary probabilities + aux loss
    chunk_kernel<<<L_TOK - 1, 256, 0, stream>>>(Qbuf, Kbuf, bpv, selv, mkv);
    aux_kernel<<<1, 256, 0, stream>>>(bpv, mkv, out + (size_t)L_TOK * VOCABN);

    // 5) boundary EMA scan -> h_final (bf16)
    scan_kernel<<<HID / 256, 256, 0, stream>>>(h, selv, mkv, hfbf);

    // 6) logits = h_final @ lm_w^T
    dim3 glm(VOCABN / 128, L_TOK / 128);
    gemm_bf16_nt_kernel<<<glm, 256, 0, stream>>>(hfbf, lmwbf, out,
                                                 L_TOK, VOCABN, HID, HID, HID, VOCABN);
}
